// DeepSeekMLP_13950053777725
// MI455X (gfx1250) — matrix operand order repaired and checked by reading, not compiled, not run on hardware
//
#include <hip/hip_runtime.h>
#include <hip/hip_bf16.h>

typedef __attribute__((ext_vector_type(16))) __bf16 v16bf;
typedef __attribute__((ext_vector_type(8)))  float  v8f;
typedef __attribute__((ext_vector_type(4)))  int    v4i;

#define AS1 __attribute__((address_space(1)))
#define AS3 __attribute__((address_space(3)))

#if defined(__has_builtin)
#  if __has_builtin(__builtin_amdgcn_global_load_async_to_lds_b128)
#    define HAVE_ASYNC_LDS 1
#  endif
#endif
#ifndef HAVE_ASYNC_LDS
#  define HAVE_ASYNC_LDS 0
#endif

namespace {

constexpr int Mdim = 4096;   // tokens
constexpr int Kdim = 7168;   // hidden size
constexpr int Edim = 2048;   // expert dim

constexpr int BM   = 128;    // block rows (m)
constexpr int BK   = 64;     // k-depth per LDS stage (2 WMMA k-steps)
constexpr int LSTR = BK + 8; // 72 bf16 row stride = 144 B (16B aligned, bank-spread)
constexpr int GSTR = 68;     // gate scratch stride in floats

constexpr int STAGE_ELEMS = BM * LSTR;            // 9216 bf16 per tile
constexpr int STAGE_BYTES = STAGE_ELEMS * 2;      // 18432 B
constexpr int SMEM_BYTES  = 4 * STAGE_BYTES;      // A0,B0,A1,B1 = 73728 B (dynamic)

struct Frag32B { uint4 lo, hi; };

__device__ __forceinline__ v16bf combine_frag(uint4 lo, uint4 hi) {
    Frag32B f{lo, hi};
    return __builtin_bit_cast(v16bf, f);
}

// A fragment (16x32 bf16): lane L holds row M=L%16; VGPR0-3 = K {0..7|8..15},
// VGPR4-7 = K {16..23|24..31} selected by lane>=16.
__device__ __forceinline__ v16bf frag_a(const __bf16* t, int row, int kb, int lane) {
    const __bf16* p = t + row * LSTR + kb + ((lane & 16) ? 8 : 0);
    return combine_frag(*(const uint4*)(p), *(const uint4*)(p + 16));
}

// B fragment (32x16 bf16): lane L holds col N=L%16; VGPR0-3 = K {0..7|8..15},
// VGPR4-7 = K {16..23|24..31} selected by lane>=16.
__device__ __forceinline__ v16bf frag_b(const __bf16* t, int row, int kb, int lane) {
    const __bf16* p = t + row * LSTR + kb + ((lane & 16) ? 8 : 0);
    return combine_frag(*(const uint4*)(p), *(const uint4*)(p + 16));
}

// one 16-byte global -> LDS chunk: async fire-and-forget when available
__device__ __forceinline__ void cp16(const __bf16* g, __bf16* l) {
#if HAVE_ASYNC_LDS
    __builtin_amdgcn_global_load_async_to_lds_b128(
        (AS1 v4i*)g, (AS3 v4i*)l, 0, 0);
#else
    *(uint4*)l = *(const uint4*)g;   // sync fallback: global_load_b128 + ds_store_b128
#endif
}

template <int N>
__device__ __forceinline__ void wait_async() {
#if HAVE_ASYNC_LDS
#  if __has_builtin(__builtin_amdgcn_s_wait_asynccnt)
    __builtin_amdgcn_s_wait_asynccnt(N);
#  else
    asm volatile("s_wait_asynccnt %0" :: "i"(N) : "memory");
#  endif
#endif
}

} // namespace

// -------------------------------------------------------------------------
// fp32 -> bf16 pre-conversion (bandwidth bound; one pass per operand)
// -------------------------------------------------------------------------
__global__ __launch_bounds__(256) void cvt_f32_bf16_kernel(
        const float* __restrict__ src, __bf16* __restrict__ dst, int n4) {
    int i = blockIdx.x * 256 + threadIdx.x;
    if (i >= n4) return;
    float4 f = ((const float4*)src)[i];
    union { __bf16 h[4]; uint2 u; } pk;
    pk.h[0] = (__bf16)f.x; pk.h[1] = (__bf16)f.y;
    pk.h[2] = (__bf16)f.z; pk.h[3] = (__bf16)f.w;
    ((uint2*)dst)[i] = pk.u;
}

// -------------------------------------------------------------------------
// Kernel 1: ug = x @ w_up_gate^T (bf16 WMMA, fp32 acc), fused SwiGLU -> h (bf16)
// Double-buffered async global->LDS pipeline (8 b128 chunks/thread/stage)
// -------------------------------------------------------------------------
__device__ __forceinline__ void prefetch_upgate(
        const __bf16* xb, const __bf16* wb, __bf16* bufA, __bf16* bufB,
        int tid, int m0, int e0, int k0) {
    #pragma unroll
    for (int i = 0; i < 4; ++i) {        // A tile: 128 x 64 bf16
        int idx = tid + i * 256;         // 0..1023 b128 chunks
        int r   = idx >> 3;
        int c   = (idx & 7) << 3;
        cp16(xb + (size_t)(m0 + r) * Kdim + k0 + c, bufA + r * LSTR + c);
    }
    #pragma unroll
    for (int i = 0; i < 4; ++i) {        // B tile: rows 0..63 gate, 64..127 up
        int idx  = tid + i * 256;
        int r    = idx >> 3;
        int c    = (idx & 7) << 3;
        int grow = (r < 64) ? (e0 + r) : (Edim + e0 + (r - 64));
        cp16(wb + (size_t)grow * Kdim + k0 + c, bufB + r * LSTR + c);
    }
}

__global__ __launch_bounds__(256) void fused_upgate_swiglu_kernel(
        const __bf16* __restrict__ xb, const __bf16* __restrict__ wb,
        __bf16* __restrict__ hbuf) {
    extern __shared__ char dynsmem[];
    __bf16* sA0 = (__bf16*)dynsmem;
    __bf16* sB0 = sA0 + STAGE_ELEMS;
    __bf16* sA1 = sB0 + STAGE_ELEMS;
    __bf16* sB1 = sA1 + STAGE_ELEMS;
    float*  gate = (float*)dynsmem;              // 128*68*4 B, reused post-loop

    const int tid  = threadIdx.x;
    const int lane = tid & 31;
    const int wave = tid >> 5;
    const int wm   = wave >> 2;                  // 0..1 : 64-row band
    const int wn   = wave & 3;                   // 0..3 : 32-col band
    const int m0   = blockIdx.y * BM;
    const int e0   = blockIdx.x * 64;            // h-column base

    v8f acc[4][2];
    #pragma unroll
    for (int i = 0; i < 4; ++i)
        #pragma unroll
        for (int j = 0; j < 2; ++j)
            acc[i][j] = v8f{0.f, 0.f, 0.f, 0.f, 0.f, 0.f, 0.f, 0.f};

    constexpr int nIter = Kdim / BK;             // 112
    prefetch_upgate(xb, wb, sA0, sB0, tid, m0, e0, 0);

    for (int it = 0; it < nIter; ++it) {
        const int cur = it & 1;
        __bf16* cA = cur ? sA1 : sA0;
        __bf16* cB = cur ? sB1 : sB0;
        if (it + 1 < nIter) {
            prefetch_upgate(xb, wb, cur ? sA0 : sA1, cur ? sB0 : sB1,
                            tid, m0, e0, (it + 1) * BK);
            wait_async<8>();                     // current stage landed
        } else {
            wait_async<0>();
        }
        __syncthreads();                         // stage visible to all waves

        #pragma unroll
        for (int kk = 0; kk < BK; kk += 32) {
            v16bf af[4], bfr[2];
            #pragma unroll
            for (int i = 0; i < 4; ++i)
                af[i] = frag_a(cA, wm * 64 + i * 16 + (lane & 15), kk, lane);
            #pragma unroll
            for (int j = 0; j < 2; ++j)
                bfr[j] = frag_b(cB, wn * 32 + j * 16 + (lane & 15), kk, lane);
            #pragma unroll
            for (int i = 0; i < 4; ++i)
                #pragma unroll
                for (int j = 0; j < 2; ++j)
                    acc[i][j] = __builtin_amdgcn_wmma_f32_16x16x32_bf16(
                        false, af[i], false, bfr[j], (short)0, acc[i][j],
                        false, false);
        }
        __syncthreads();                         // reads done before overwrite
    }

    // ---- fused SwiGLU epilogue: h[m][e] = silu(gate)*up ----
    if (wn < 2) {                                // gate waves: cols 0..63 -> LDS
        #pragma unroll
        for (int i = 0; i < 4; ++i)
            #pragma unroll
            for (int j = 0; j < 2; ++j)
                #pragma unroll
                for (int v = 0; v < 8; ++v) {
                    int row = wm * 64 + i * 16 + v + ((lane & 16) ? 8 : 0);
                    int col = wn * 32 + j * 16 + (lane & 15);
                    gate[row * GSTR + col] = acc[i][j][v];
                }
    }
    __syncthreads();
    if (wn >= 2) {                               // up waves pair with gate cols
        #pragma unroll
        for (int i = 0; i < 4; ++i)
            #pragma unroll
            for (int j = 0; j < 2; ++j)
                #pragma unroll
                for (int v = 0; v < 8; ++v) {
                    int row = wm * 64 + i * 16 + v + ((lane & 16) ? 8 : 0);
                    int e   = (wn - 2) * 32 + j * 16 + (lane & 15);
                    float g = gate[row * GSTR + e];
                    float u = acc[i][j][v];
                    float hv = (g / (1.0f + __expf(-g))) * u;  // silu(g)*u
                    hbuf[(size_t)(m0 + row) * Edim + e0 + e] = (__bf16)hv;
                }
    }
}

// -------------------------------------------------------------------------
// Kernel 2: out = h @ w_down^T  (h bf16 [M,E], w_down bf16 [Kdim,E], out fp32)
// -------------------------------------------------------------------------
__device__ __forceinline__ void prefetch_down(
        const __bf16* hb, const __bf16* wb, __bf16* bufA, __bf16* bufB,
        int tid, int m0, int n0, int k0) {
    #pragma unroll
    for (int i = 0; i < 4; ++i) {
        int idx = tid + i * 256;
        int r   = idx >> 3;
        int c   = (idx & 7) << 3;
        cp16(hb + (size_t)(m0 + r) * Edim + k0 + c, bufA + r * LSTR + c);
    }
    #pragma unroll
    for (int i = 0; i < 4; ++i) {
        int idx = tid + i * 256;
        int r   = idx >> 3;
        int c   = (idx & 7) << 3;
        cp16(wb + (size_t)(n0 + r) * Edim + k0 + c, bufB + r * LSTR + c);
    }
}

__global__ __launch_bounds__(256) void down_gemm_kernel(
        const __bf16* __restrict__ hbuf, const __bf16* __restrict__ wdnb,
        float* __restrict__ out) {
    extern __shared__ char dynsmem[];
    __bf16* sA0 = (__bf16*)dynsmem;
    __bf16* sB0 = sA0 + STAGE_ELEMS;
    __bf16* sA1 = sB0 + STAGE_ELEMS;
    __bf16* sB1 = sA1 + STAGE_ELEMS;

    const int tid  = threadIdx.x;
    const int lane = tid & 31;
    const int wave = tid >> 5;
    const int wm   = wave >> 2;
    const int wn   = wave & 3;
    const int m0   = blockIdx.y * BM;
    const int n0   = blockIdx.x * 128;

    v8f acc[4][2];
    #pragma unroll
    for (int i = 0; i < 4; ++i)
        #pragma unroll
        for (int j = 0; j < 2; ++j)
            acc[i][j] = v8f{0.f, 0.f, 0.f, 0.f, 0.f, 0.f, 0.f, 0.f};

    constexpr int nIter = Edim / BK;             // 32
    prefetch_down(hbuf, wdnb, sA0, sB0, tid, m0, n0, 0);

    for (int it = 0; it < nIter; ++it) {
        const int cur = it & 1;
        __bf16* cA = cur ? sA1 : sA0;
        __bf16* cB = cur ? sB1 : sB0;
        if (it + 1 < nIter) {
            prefetch_down(hbuf, wdnb, cur ? sA0 : sA1, cur ? sB0 : sB1,
                          tid, m0, n0, (it + 1) * BK);
            wait_async<8>();
        } else {
            wait_async<0>();
        }
        __syncthreads();

        #pragma unroll
        for (int kk = 0; kk < BK; kk += 32) {
            v16bf af[4], bfr[2];
            #pragma unroll
            for (int i = 0; i < 4; ++i)
                af[i] = frag_a(cA, wm * 64 + i * 16 + (lane & 15), kk, lane);
            #pragma unroll
            for (int j = 0; j < 2; ++j)
                bfr[j] = frag_b(cB, wn * 32 + j * 16 + (lane & 15), kk, lane);
            #pragma unroll
            for (int i = 0; i < 4; ++i)
                #pragma unroll
                for (int j = 0; j < 2; ++j)
                    acc[i][j] = __builtin_amdgcn_wmma_f32_16x16x32_bf16(
                        false, af[i], false, bfr[j], (short)0, acc[i][j],
                        false, false);
        }
        __syncthreads();
    }

    // epilogue: fp32 store; C layout: VGPR v -> M = v + (lane>=16 ? 8:0), N = lane%16
    #pragma unroll
    for (int i = 0; i < 4; ++i)
        #pragma unroll
        for (int j = 0; j < 2; ++j)
            #pragma unroll
            for (int v = 0; v < 8; ++v) {
                int row = wm * 64 + i * 16 + v + ((lane & 16) ? 8 : 0);
                int col = wn * 32 + j * 16 + (lane & 15);
                out[(size_t)(m0 + row) * Kdim + n0 + col] = acc[i][j][v];
            }
}

extern "C" void kernel_launch(void* const* d_in, const int* in_sizes, int n_in,
                              void* d_out, int out_size, void* d_ws, size_t ws_size,
                              hipStream_t stream) {
    (void)in_sizes; (void)n_in; (void)out_size; (void)ws_size;
    const float* x   = (const float*)d_in[0];   // [4096, 7168]
    const float* wug = (const float*)d_in[1];   // [4096, 7168]  (2E rows, k-major)
    const float* wdn = (const float*)d_in[2];   // [7168, 2048]  (k rows, e-major)
    float* out       = (float*)d_out;           // [4096, 7168]

    // workspace layout (bf16): x | w_up_gate | w_down | h   = ~164 MB
    __bf16* xb   = (__bf16*)d_ws;
    __bf16* wugb = xb   + (size_t)Mdim * Kdim;
    __bf16* wdnb = wugb + (size_t)2 * Edim * Kdim;
    __bf16* hbuf = wdnb + (size_t)Kdim * Edim;

    dim3 blk(256);
    {   // fp32 -> bf16 pre-conversion passes
        int n4x = (Mdim * Kdim) / 4;            // 7,340,032
        int n4w = (2 * Edim * Kdim) / 4;        // 7,340,032
        int n4d = (Kdim * Edim) / 4;            // 3,670,016
        cvt_f32_bf16_kernel<<<dim3((n4x + 255) / 256), blk, 0, stream>>>(x,   xb,   n4x);
        cvt_f32_bf16_kernel<<<dim3((n4w + 255) / 256), blk, 0, stream>>>(wug, wugb, n4w);
        cvt_f32_bf16_kernel<<<dim3((n4d + 255) / 256), blk, 0, stream>>>(wdn, wdnb, n4d);
    }

    dim3 g1(Edim / 64, Mdim / BM);              // 32 x 32
    dim3 g2(Kdim / 128, Mdim / BM);             // 56 x 32
    fused_upgate_swiglu_kernel<<<g1, blk, SMEM_BYTES, stream>>>(xb, wugb, hbuf);
    down_gemm_kernel<<<g2, blk, SMEM_BYTES, stream>>>(hbuf, wdnb, out);
}
